// GIN_Model_79680233276313
// MI455X (gfx1250) — matrix operand order repaired and checked by reading, not compiled, not run on hardware
//
#include <hip/hip_runtime.h>

// ---------------- problem constants (match reference) ----------------
#define N_NODES  50000
#define NPAD     50048          // 391 * 128, padded so WMMA tiles have full EXEC
#define DD       128
#define NLAYERS  5
#define NEDGES   800000
#define NGRAPHS  512
#define NCLASS   10
#define MERGED_K 768            // (L+1)*D

// dynamic LDS layout for gin_mlp_kernel
#define LDS_W1   0
#define LDS_W2   32768
#define LDS_H1   65536          // 8 waves * 4096B (16x128 bf16 tile each)
#define LDS_B1   98304
#define LDS_B2   98816
#define LDS_TOT  99328

typedef __bf16 bf16;
typedef __attribute__((ext_vector_type(16))) __bf16 v16bf;
typedef __attribute__((ext_vector_type(8)))  float  v8f;

// ---------------- WMMA helpers (CDNA5 wave32, 16x16x32 bf16) ----------------
__device__ __forceinline__ v8f wmma_bf16(v16bf a, v16bf b, v8f c) {
    // (neg_a, A, neg_b, B, c_mod, C, reuse_a, reuse_b)
    return __builtin_amdgcn_wmma_f32_16x16x32_bf16(false, a, false, b, (short)0, c, false, false);
}

// A-matrix 16x32 bf16 fragment from row-major [16 x ld] source.
// lanes 0-15: M=lane, K = kb+{0..7} and kb+{16..23}
// lanes16-31: M=lane-16, K = kb+{8..15} and kb+{24..31}
__device__ __forceinline__ v16bf load_fragA(const bf16* __restrict__ base, int ld, int kb, int lane) {
    int m = lane & 15;
    int kOff = kb + ((lane >> 4) << 3);
    const bf16* p = base + (size_t)m * ld + kOff;
    union { v16bf v; uint4 q[2]; } u;
    u.q[0] = *(const uint4*)(p);
    u.q[1] = *(const uint4*)(p + 16);
    return u.v;
}

// B-matrix 32x16 bf16 fragment; source is transposed weights Wt[out][in] row-major.
// lanes 0-15: N=lane, K = kb+{0..7} and kb+{16..23} ; lanes 16-31: N=lane-16, K = kb+{8..15} and kb+{24..31}
__device__ __forceinline__ v16bf load_fragB(const bf16* __restrict__ baseT, int ld, int kb, int lane) {
    int n = lane & 15;
    int kOff = kb + ((lane >> 4) << 3);
    const bf16* p = baseT + (size_t)n * ld + kOff;
    union { v16bf v; uint4 q[2]; } u;
    u.q[0] = *(const uint4*)(p);
    u.q[1] = *(const uint4*)(p + 16);
    return u.v;
}

// ---------------- small utility kernels ----------------
__global__ void zero_f32(float* __restrict__ p, int n) {
    int t = blockIdx.x * blockDim.x + threadIdx.x;
    if (t < n) p[t] = 0.f;
}

// dst[b][o][i] = (bf16) src[b][i][o]   (src: [batch, rows(in), cols(out)])
__global__ void transpose_to_bf16(const float* __restrict__ src, bf16* __restrict__ dst,
                                  int batch, int rows, int cols) {
    int t = blockIdx.x * blockDim.x + threadIdx.x;
    int total = batch * rows * cols;
    if (t >= total) return;
    int b   = t / (rows * cols);
    int rem = t - b * rows * cols;
    int o   = rem / rows;
    int i   = rem - o * rows;
    dst[t] = (bf16)src[(size_t)(b * rows + i) * cols + o];
}

// h = x (padded rows -> 0); pool0[batch[row]] += x[row]
__global__ void init_h(const float* __restrict__ x, float* __restrict__ h,
                       float* __restrict__ pool0, const int* __restrict__ batch,
                       int nValid, int total) {
    int t = blockIdx.x * blockDim.x + threadIdx.x;
    if (t >= total) return;
    int row = t >> 7;
    float v = (row < nValid) ? x[t] : 0.f;
    h[t] = v;
    if (row < nValid) atomicAdd(pool0 + (size_t)batch[row] * DD + (t & 127), v);
}

// agg[src] += mask * h[dst] ; 32 lanes per edge, float4 per lane (coalesced 512B row)
__global__ void edge_agg(const int* __restrict__ src, const int* __restrict__ dst,
                         const float* __restrict__ mask, const float* __restrict__ h,
                         float* __restrict__ agg) {
    int t = blockIdx.x * blockDim.x + threadIdx.x;
    int e = t >> 5;
    if (e >= NEDGES) return;
    int c = (t & 31) << 2;
    int s = src[e], d = dst[e];
    float m = mask[e];
    const float4 v = *(const float4*)(h + (size_t)d * DD + c);
    float* o = agg + (size_t)s * DD + c;
    atomicAdd(o + 0, m * v.x);
    atomicAdd(o + 1, m * v.y);
    atomicAdd(o + 2, m * v.z);
    atomicAdd(o + 3, m * v.w);
}

// aBf = bf16( agg + (1+eps[l]) * h )
__global__ void pooled_to_bf16(const float* __restrict__ agg, const float* __restrict__ h,
                               const float* __restrict__ eps, int l,
                               bf16* __restrict__ outB, int total) {
    int t = blockIdx.x * blockDim.x + threadIdx.x;
    if (t >= total) return;
    float s = 1.0f + eps[l];
    outB[t] = (bf16)(agg[t] + s * h[t]);
}

// ---------------- fused GIN MLP: relu(relu(A@W1+b1)@W2+b2), + graph pooling ----------------
// block = 256 threads = 8 waves; each wave owns a 16-row node tile; 128 rows per block.
// Weights are staged into LDS with CDNA5 async loads (ASYNCcnt path).
__global__ void __launch_bounds__(256)
gin_mlp_kernel(const bf16* __restrict__ aIn,     // [NPAD,128] bf16 (pooled features)
               const bf16* __restrict__ w1t,     // [128 out][128 in] bf16
               const bf16* __restrict__ w2t,     // [128 out][128 in] bf16
               const float* __restrict__ b1, const float* __restrict__ b2,
               float* __restrict__ hOut,         // [NPAD,128] f32
               float* __restrict__ poolOut,      // [G,128] f32 (atomic accumulate)
               const int* __restrict__ batch, int nValid) {
    extern __shared__ __align__(16) char smem[];
    bf16*  sW1 = (bf16*)(smem + LDS_W1);
    bf16*  sW2 = (bf16*)(smem + LDS_W2);
    float* sB1 = (float*)(smem + LDS_B1);
    float* sB2 = (float*)(smem + LDS_B2);

    int tid  = threadIdx.x;
    int wave = tid >> 5, lane = tid & 31;
    bf16* tile = (bf16*)(smem + LDS_H1 + wave * 4096);   // this wave's 16x128 h1 tile

    // ---- async-stage both weight matrices (32KB each) into LDS ----
    {
        unsigned l1 = (unsigned)(size_t)sW1;
        unsigned l2 = (unsigned)(size_t)sW2;
        const char* g1 = (const char*)w1t;
        const char* g2 = (const char*)w2t;
#pragma unroll
        for (int it = 0; it < 8; ++it) {
            int off = tid * 16 + it * 4096;
            unsigned d1 = l1 + off;
            unsigned d2 = l2 + off;
            const char* a1 = g1 + off;
            const char* a2 = g2 + off;
            asm volatile("global_load_async_to_lds_b128 %0, %1, off"
                         :: "v"(d1), "v"(a1) : "memory");
            asm volatile("global_load_async_to_lds_b128 %0, %1, off"
                         :: "v"(d2), "v"(a2) : "memory");
        }
    }
    if (tid < 128) { sB1[tid] = b1[tid]; sB2[tid] = b2[tid]; }
    asm volatile("s_wait_asynccnt 0x0" ::: "memory");
    __syncthreads();

    int rowBase = blockIdx.x * 128 + wave * 16;
    const bf16* aRow = aIn + (size_t)rowBase * DD;
    int n0 = lane & 15;
    int r0 = (lane >> 4) << 3;

    const v8f vzero = {0.f, 0.f, 0.f, 0.f, 0.f, 0.f, 0.f, 0.f};
    v8f acc[8];
#pragma unroll
    for (int i = 0; i < 8; ++i) acc[i] = vzero;

    // ---- matmul1: h1 = A @ W1 (B fragments from LDS) ----
#pragma unroll
    for (int kb = 0; kb < 128; kb += 32) {
        v16bf aF = load_fragA(aRow, DD, kb, lane);
#pragma unroll
        for (int nt = 0; nt < 8; ++nt) {
            v16bf bF = load_fragB(sW1 + (size_t)nt * 16 * DD, DD, kb, lane);
            acc[nt] = wmma_bf16(aF, bF, acc[nt]);
        }
    }

    // bias + relu, stage to LDS as bf16 row-major [16][128]
#pragma unroll
    for (int nt = 0; nt < 8; ++nt) {
#pragma unroll
        for (int v = 0; v < 8; ++v) {
            float val = fmaxf(acc[nt][v] + sB1[nt * 16 + n0], 0.f);
            tile[(r0 + v) * DD + nt * 16 + n0] = (bf16)val;
        }
    }
    __syncthreads();

    // ---- matmul2: h = h1 @ W2 (A from LDS tile, B from LDS weights) ----
#pragma unroll
    for (int i = 0; i < 8; ++i) acc[i] = vzero;
#pragma unroll
    for (int kb = 0; kb < 128; kb += 32) {
        v16bf aF = load_fragA(tile, DD, kb, lane);
#pragma unroll
        for (int nt = 0; nt < 8; ++nt) {
            v16bf bF = load_fragB(sW2 + (size_t)nt * 16 * DD, DD, kb, lane);
            acc[nt] = wmma_bf16(aF, bF, acc[nt]);
        }
    }

    // bias + relu -> hOut (f32), and atomic graph pooling for valid rows
#pragma unroll
    for (int v = 0; v < 8; ++v) {
        int row = rowBase + r0 + v;
        int g = (row < nValid) ? batch[row] : -1;
        float* hrow = hOut + (size_t)row * DD;
#pragma unroll
        for (int nt = 0; nt < 8; ++nt) {
            float val = fmaxf(acc[nt][v] + sB2[nt * 16 + n0], 0.f);
            hrow[nt * 16 + n0] = val;
            if (g >= 0) atomicAdd(poolOut + (size_t)g * DD + nt * 16 + n0, val);
        }
    }
}

// merged[g][l*128+d] = bf16( poolG[l][g][d] )
__global__ void build_merged(const float* __restrict__ poolG, bf16* __restrict__ merged) {
    int t = blockIdx.x * blockDim.x + threadIdx.x;
    if (t >= NGRAPHS * MERGED_K) return;
    int g = t / MERGED_K;
    int col = t - g * MERGED_K;
    int l = col >> 7;
    int d = col & 127;
    merged[t] = (bf16)poolG[((size_t)l * NGRAPHS + g) * DD + d];
}

// lin1: relu(merged[512,768] @ W[768,768] + b) via WMMA; one 16x16 tile per wave
__global__ void __launch_bounds__(256)
lin1_gemm(const bf16* __restrict__ A,       // [512,768]
          const bf16* __restrict__ Bt,      // [768 out][768 in]
          const float* __restrict__ bias,
          bf16* __restrict__ out) {         // [512,768]
    int wave = threadIdx.x >> 5, lane = threadIdx.x & 31;
    int w = blockIdx.x * 8 + wave;          // 32*48 = 1536 tiles
    int mT = w & 31;
    int nT = w >> 5;
    const bf16* aBase = A  + (size_t)mT * 16 * MERGED_K;
    const bf16* bBase = Bt + (size_t)nT * 16 * MERGED_K;

    v8f acc = {0.f, 0.f, 0.f, 0.f, 0.f, 0.f, 0.f, 0.f};
#pragma unroll 4
    for (int kb = 0; kb < MERGED_K; kb += 32) {
        v16bf aF = load_fragA(aBase, MERGED_K, kb, lane);
        v16bf bF = load_fragB(bBase, MERGED_K, kb, lane);
        acc = wmma_bf16(aF, bF, acc);
    }
    int n0 = lane & 15;
    int r0 = (lane >> 4) << 3;
#pragma unroll
    for (int v = 0; v < 8; ++v) {
        float val = fmaxf(acc[v] + bias[nT * 16 + n0], 0.f);
        out[(size_t)(mT * 16 + r0 + v) * MERGED_K + nT * 16 + n0] = (bf16)val;
    }
}

// lin2 (768 -> 10) + softmax; one wave per graph, lane c < 10 owns class c
__global__ void __launch_bounds__(32)
head_kernel(const bf16* __restrict__ lin1o, const float* __restrict__ w2,
            const float* __restrict__ b2, float* __restrict__ out) {
    int g = blockIdx.x;
    int lane = threadIdx.x;
    float dot = 0.f;
    if (lane < NCLASS) {
        dot = b2[lane];
        const bf16* row = lin1o + (size_t)g * MERGED_K;
        for (int i = 0; i < MERGED_K; ++i)
            dot += (float)row[i] * w2[(size_t)i * NCLASS + lane];
    }
    float x = (lane < NCLASS) ? dot : -3.4e38f;
#pragma unroll
    for (int off = 16; off; off >>= 1) x = fmaxf(x, __shfl_xor(x, off));
    float e = (lane < NCLASS) ? __expf(dot - x) : 0.f;
    float s = e;
#pragma unroll
    for (int off = 16; off; off >>= 1) s += __shfl_xor(s, off);
    if (lane < NCLASS) out[g * NCLASS + lane] = e / s;
}

// ---------------- launch ----------------
extern "C" void kernel_launch(void* const* d_in, const int* in_sizes, int n_in,
                              void* d_out, int out_size, void* d_ws, size_t ws_size,
                              hipStream_t stream) {
    (void)in_sizes; (void)n_in; (void)out_size; (void)ws_size;
    const float* x     = (const float*)d_in[0];
    const int*   eidx  = (const int*)d_in[1];   // [2,E]
    const float* emask = (const float*)d_in[2];
    const int*   batch = (const int*)d_in[3];
    // d_in[4] = num_graphs (scalar, known constant 512)
    const float* eps   = (const float*)d_in[5];
    const float* w1    = (const float*)d_in[6];
    const float* b1    = (const float*)d_in[7];
    const float* w2    = (const float*)d_in[8];
    const float* b2    = (const float*)d_in[9];
    const float* lin1w = (const float*)d_in[10];
    const float* lin1b = (const float*)d_in[11];
    const float* lin2w = (const float*)d_in[12];
    const float* lin2b = (const float*)d_in[13];
    float* out = (float*)d_out;

    // workspace carve-up (~69 MB total)
    char* wsp = (char*)d_ws;
    auto carve = [&](size_t bytes) { char* p = wsp; wsp += (bytes + 255) & ~(size_t)255; return p; };
    float* hCur   = (float*)carve((size_t)NPAD * DD * 4);
    float* agg    = (float*)carve((size_t)NPAD * DD * 4);
    bf16*  aBf    = (bf16*) carve((size_t)NPAD * DD * 2);
    float* poolG  = (float*)carve((size_t)(NLAYERS + 1) * NGRAPHS * DD * 4);
    bf16*  w1t    = (bf16*) carve((size_t)NLAYERS * DD * DD * 2);
    bf16*  w2t    = (bf16*) carve((size_t)NLAYERS * DD * DD * 2);
    bf16*  lin1t  = (bf16*) carve((size_t)MERGED_K * MERGED_K * 2);
    bf16*  merged = (bf16*) carve((size_t)NGRAPHS * MERGED_K * 2);
    bf16*  lin1o  = (bf16*) carve((size_t)NGRAPHS * MERGED_K * 2);

    const int* srcI = eidx;
    const int* dstI = eidx + NEDGES;

    const int hTot  = NPAD * DD;
    const int poolN = (NLAYERS + 1) * NGRAPHS * DD;

    zero_f32<<<(poolN + 255) / 256, 256, 0, stream>>>(poolG, poolN);
    transpose_to_bf16<<<(NLAYERS * DD * DD + 255) / 256, 256, 0, stream>>>(w1, w1t, NLAYERS, DD, DD);
    transpose_to_bf16<<<(NLAYERS * DD * DD + 255) / 256, 256, 0, stream>>>(w2, w2t, NLAYERS, DD, DD);
    transpose_to_bf16<<<(MERGED_K * MERGED_K + 255) / 256, 256, 0, stream>>>(lin1w, lin1t, 1, MERGED_K, MERGED_K);
    init_h<<<(hTot + 255) / 256, 256, 0, stream>>>(x, hCur, poolG, batch, N_NODES, hTot);

    for (int l = 0; l < NLAYERS; ++l) {
        zero_f32<<<(hTot + 255) / 256, 256, 0, stream>>>(agg, hTot);
        edge_agg<<<(NEDGES * 32) / 256, 256, 0, stream>>>(srcI, dstI, emask, hCur, agg);
        pooled_to_bf16<<<(hTot + 255) / 256, 256, 0, stream>>>(agg, hCur, eps, l, aBf, hTot);
        gin_mlp_kernel<<<NPAD / 128, 256, LDS_TOT, stream>>>(
            aBf, w1t + (size_t)l * DD * DD, w2t + (size_t)l * DD * DD,
            b1 + (size_t)l * DD, b2 + (size_t)l * DD,
            hCur, poolG + (size_t)(l + 1) * NGRAPHS * DD, batch, N_NODES);
    }

    build_merged<<<(NGRAPHS * MERGED_K + 255) / 256, 256, 0, stream>>>(poolG, merged);
    lin1_gemm<<<(32 * 48) / 8, 256, 0, stream>>>(merged, lin1t, lin1b, lin1o);
    head_kernel<<<NGRAPHS, 32, 0, stream>>>(lin1o, lin2w, lin2b, out);
}
